// Decoder_17952963298115
// MI455X (gfx1250) — matrix operand order repaired and checked by reading, not compiled, not run on hardware
//
#include <hip/hip_runtime.h>

// ---------------------------------------------------------------------------
// x[131072, 512] fp32, W[512,512] fp32
//  0) split W -> bf16 hi/lo planes (once, in ws; L2-resident)
//  1) GEMM y = x*W^T via bf16x3 on V_WMMA_F32_16X16X32_BF16
//     - A tile: global->reg->cvt_pk(bf16 hi/lo)->LDS
//     - B tile: GLOBAL_LOAD_ASYNC_TO_LDS_B128, double-buffered (pure copy)
//  2) BN batch stats + fold gamma/beta
//  3) LIF over T=4 in place on d_out
// ---------------------------------------------------------------------------
#define MROWS   131072
#define KDIM    512
#define NDIM    512
#define TSTEPS  4
#define PER_T   16777216

#define BM 128
#define BN 128
#define BK 32
#define LDA2 (BK + 8)      // bf16 stride 40 elems = 80B: rows 16B-aligned, 20-word
                           // lane stride -> conflict-free LDS banks

typedef __bf16 v16bf __attribute__((ext_vector_type(16)));
typedef __bf16 bf2   __attribute__((ext_vector_type(2)));
typedef float  v8f   __attribute__((ext_vector_type(8)));
typedef int    v4i   __attribute__((ext_vector_type(4)));

typedef __attribute__((address_space(1))) v4i* gptr_v4i;
typedef __attribute__((address_space(3))) v4i* lptr_v4i;

#define WS_WH_OFF 8192     // ws bytes: [0,8K) stats; [8K,+512K) Wh; then Wl

// ---------------- CDNA5 async global->LDS copy (ASYNCcnt-tracked) ----------
__device__ __forceinline__ void async_b128(const unsigned short* gsrc,
                                           unsigned short* ldst)
{
#if __has_builtin(__builtin_amdgcn_global_load_async_to_lds_b128)
    __builtin_amdgcn_global_load_async_to_lds_b128(
        (gptr_v4i)(void*)gsrc, (lptr_v4i)(void*)ldst, 0, 0);
#else
    unsigned lds_off = (unsigned)(unsigned long long)
        (__attribute__((address_space(3))) void*)ldst;
    asm volatile("global_load_async_to_lds_b128 %0, %1, off"
                 :: "v"(lds_off), "v"(gsrc) : "memory");
#endif
}
__device__ __forceinline__ void wait_async_le4() {
#if __has_builtin(__builtin_amdgcn_s_wait_asynccnt)
    __builtin_amdgcn_s_wait_asynccnt(4);
#else
    asm volatile("s_wait_asynccnt 0x4" ::: "memory");
#endif
}
__device__ __forceinline__ void wait_async_le0() {
#if __has_builtin(__builtin_amdgcn_s_wait_asynccnt)
    __builtin_amdgcn_s_wait_asynccnt(0);
#else
    asm volatile("s_wait_asynccnt 0x0" ::: "memory");
#endif
}

// ------------------------- bf16 split helpers ------------------------------
// packed f32x2 -> bf16x2 (RNE) as a 32-bit word
__device__ __forceinline__ unsigned cvt_pk_bf16(float x, float y) {
#if __has_builtin(__builtin_amdgcn_cvt_pk_bf16_f32)
    auto h = __builtin_amdgcn_cvt_pk_bf16_f32(x, y);
    return __builtin_bit_cast(unsigned, h);
#else
    bf2 h;
    h.x = (__bf16)x;
    h.y = (__bf16)y;
    return __builtin_bit_cast(unsigned, h);
#endif
}
// split two floats into packed bf16 hi + lo words
// (bf16 -> f32 widening is exact: just a 16-bit shift / mask)
__device__ __forceinline__ void split2(float x, float y, unsigned& hi, unsigned& lo) {
    const unsigned h  = cvt_pk_bf16(x, y);
    const float    hx = __uint_as_float(h << 16);
    const float    hy = __uint_as_float(h & 0xFFFF0000u);
    lo = cvt_pk_bf16(x - hx, y - hy);
    hi = h;
}
__device__ __forceinline__ void split4(const float4 f, uint2& hi, uint2& lo) {
    split2(f.x, f.y, hi.x, lo.x);
    split2(f.z, f.w, hi.y, lo.y);
}

// fragment loads from LDS, per ISA 7.12.2 16-bit A(16x32)/B(32x16) layouts
__device__ __forceinline__ v16bf ldfragA(const unsigned short* row, int half) {
    union { uint4 q[2]; v16bf v; } u;
    u.q[0] = *(const uint4*)(row + 8 * half);        // K = 8*half .. +7
    u.q[1] = *(const uint4*)(row + 16 + 8 * half);   // K = 16+8*half .. +7
    return u.v;
}
__device__ __forceinline__ v16bf ldfragB(const unsigned short* row, int half) {
    union { uint4 q[2]; v16bf v; } u;
    u.q[0] = *(const uint4*)(row + 8 * half);        // K = 8*half .. +7 (col n)
    u.q[1] = *(const uint4*)(row + 16 + 8 * half);   // K = 16+8*half .. +7
    return u.v;
}

#define WMMA_BF16(A, B, C) \
    __builtin_amdgcn_wmma_f32_16x16x32_bf16(false, (A), false, (B), (short)0, (C), false, false)

// ---------------------------------------------------------------------------
// Kernel 0: one-time W -> bf16 hi/lo planes
// ---------------------------------------------------------------------------
__global__ __launch_bounds__(256) void split_w_kernel(
    const float* __restrict__ W,
    unsigned short* __restrict__ Wh, unsigned short* __restrict__ Wl)
{
    const int i = blockIdx.x * 256 + threadIdx.x;    // < 262144
    const float w = W[i];
    const __bf16 h = (__bf16)w;
    const __bf16 l = (__bf16)(w - (float)h);
    Wh[i] = __builtin_bit_cast(unsigned short, h);
    Wl[i] = __builtin_bit_cast(unsigned short, l);
}

// ---------------------------------------------------------------------------
// Kernel 1: GEMM bf16x3, grid = (4, 1024), 256 thr = 8 waves (2M x 4N)
//   wave tile 64x32 -> acc[4][2]; BK=32; B tiles double-buffered via async DMA
// ---------------------------------------------------------------------------
__global__ __launch_bounds__(256) void gemm_bf16x3_wmma(
    const float* __restrict__ X,
    const unsigned short* __restrict__ Wh, const unsigned short* __restrict__ Wl,
    float* __restrict__ Y)
{
    __shared__ __attribute__((aligned(16))) unsigned short AhS[BM * LDA2];
    __shared__ __attribute__((aligned(16))) unsigned short AlS[BM * LDA2];
    __shared__ __attribute__((aligned(16))) unsigned short BhS[2][BN * LDA2];
    __shared__ __attribute__((aligned(16))) unsigned short BlS[2][BN * LDA2];

    const int tid  = threadIdx.x;
    const int lane = tid & 31;
    const int wave = tid >> 5;
    const int half = lane >> 4;
    const int lm   = lane & 15;

    const int wmo = (wave >> 2) * 64;
    const int wno = (wave & 3) * 32;

    const long m0 = (long)blockIdx.y * BM;
    const int  n0 = blockIdx.x * BN;

    // staging map: 2 threads per row, each thread owns 16 elements of the row
    const int ar = tid >> 1;            // row 0..127
    const int ac = (tid & 1) * 16;      // element offset 0 or 16

    const float*          Aptr = X  + (m0 + ar) * (long)KDIM + ac;
    const unsigned short* BhP  = Wh + (long)(n0 + ar) * KDIM + ac;
    const unsigned short* BlP  = Wl + (long)(n0 + ar) * KDIM + ac;
    const int ldsRow = ar * LDA2 + ac;

    // chunk 0: A into registers, B async into buffer 0 (4 x b128 per thread)
    float4 raf0 = *(const float4*)(Aptr + 0);
    float4 raf1 = *(const float4*)(Aptr + 4);
    float4 raf2 = *(const float4*)(Aptr + 8);
    float4 raf3 = *(const float4*)(Aptr + 12);
    async_b128(BhP,     &BhS[0][ldsRow]);
    async_b128(BhP + 8, &BhS[0][ldsRow + 8]);
    async_b128(BlP,     &BlS[0][ldsRow]);
    async_b128(BlP + 8, &BlS[0][ldsRow + 8]);

    v8f acc[4][2] = {};

    const int nChunks = KDIM / BK;      // 16
    for (int kc = 0; kc < nChunks; ++kc) {
        const int cur = kc & 1;
        const int nxt = cur ^ 1;

        uint2 h0, l0, h1, l1, h2, l2, h3, l3;
        split4(raf0, h0, l0); split4(raf1, h1, l1);
        split4(raf2, h2, l2); split4(raf3, h3, l3);

        __syncthreads();   // everyone done reading A LDS and B buf[nxt]
        *(uint4*)&AhS[ldsRow]     = make_uint4(h0.x, h0.y, h1.x, h1.y);
        *(uint4*)&AhS[ldsRow + 8] = make_uint4(h2.x, h2.y, h3.x, h3.y);
        *(uint4*)&AlS[ldsRow]     = make_uint4(l0.x, l0.y, l1.x, l1.y);
        *(uint4*)&AlS[ldsRow + 8] = make_uint4(l2.x, l2.y, l3.x, l3.y);

        if (kc + 1 < nChunks) {
            const int ko = (kc + 1) * BK;
            async_b128(BhP + ko,     &BhS[nxt][ldsRow]);      // chunk kc+1 in flight
            async_b128(BhP + ko + 8, &BhS[nxt][ldsRow + 8]);
            async_b128(BlP + ko,     &BlS[nxt][ldsRow]);
            async_b128(BlP + ko + 8, &BlS[nxt][ldsRow + 8]);
            const float* An = Aptr + ko;
            raf0 = *(const float4*)(An + 0);
            raf1 = *(const float4*)(An + 4);
            raf2 = *(const float4*)(An + 8);
            raf3 = *(const float4*)(An + 12);
            if (kc + 2 < nChunks)
                __builtin_prefetch((const void*)(Aptr + (kc + 2) * BK), 0, 3);
            wait_async_le4();   // async completes in order: oldest 4 (chunk kc) done
        } else {
            wait_async_le0();   // final chunk: drain
        }
        __syncthreads();        // A stores + all waves' chunk-kc B data visible

        v16bf fah[4], fal[4];
        #pragma unroll
        for (int i = 0; i < 4; ++i) {
            fah[i] = ldfragA(&AhS[(wmo + 16 * i + lm) * LDA2], half);
            fal[i] = ldfragA(&AlS[(wmo + 16 * i + lm) * LDA2], half);
        }
        #pragma unroll
        for (int j = 0; j < 2; ++j) {
            const v16bf fbh = ldfragB(&BhS[cur][(wno + 16 * j + lm) * LDA2], half);
            const v16bf fbl = ldfragB(&BlS[cur][(wno + 16 * j + lm) * LDA2], half);
            #pragma unroll
            for (int i = 0; i < 4; ++i) {
                acc[i][j] = WMMA_BF16(fah[i], fbh, acc[i][j]);   // hi*hi
                acc[i][j] = WMMA_BF16(fah[i], fbl, acc[i][j]);   // hi*lo
                acc[i][j] = WMMA_BF16(fal[i], fbh, acc[i][j]);   // lo*hi
            }
        }
    }

    // C/D layout: VGPR r -> M = r + 8*half, N = lm
    #pragma unroll
    for (int i = 0; i < 4; ++i) {
        #pragma unroll
        for (int j = 0; j < 2; ++j) {
            const long mB = m0 + wmo + 16 * i + 8 * half;
            const int  nO = n0 + wno + 16 * j + lm;
            #pragma unroll
            for (int r = 0; r < 8; ++r)
                Y[(mB + r) * NDIM + nO] = acc[i][j][r];
        }
    }
}

// ---------------------------------------------------------------------------
// Kernel 2: per-channel sum / sumsq of Y (ws[0..1023] pre-zeroed)
// ---------------------------------------------------------------------------
__global__ __launch_bounds__(256) void stats_kernel(
    const float* __restrict__ Y, float* __restrict__ ws)
{
    const int t  = threadIdx.x;
    const int c0 = t, c1 = t + 256;
    const long rowBase = (long)blockIdx.x * 128;
    float s0 = 0.f, q0 = 0.f, s1 = 0.f, q1 = 0.f;
    for (int r = 0; r < 128; ++r) {
        const float* row = Y + (rowBase + r) * (long)NDIM;
        const float v0 = row[c0];
        const float v1 = row[c1];
        s0 += v0; q0 += v0 * v0;
        s1 += v1; q1 += v1 * v1;
    }
    atomicAdd(&ws[c0],       s0);
    atomicAdd(&ws[512 + c0], q0);
    atomicAdd(&ws[c1],       s1);
    atomicAdd(&ws[512 + c1], q1);
}

// ---------------------------------------------------------------------------
// Kernel 3: fold BN into per-channel scale/shift
// ---------------------------------------------------------------------------
__global__ __launch_bounds__(256) void finalize_kernel(
    const float* __restrict__ gamma, const float* __restrict__ beta,
    float* __restrict__ ws)
{
    const int o = blockIdx.x * 256 + threadIdx.x;
    if (o < NDIM) {
        const float invM = 1.0f / (float)MROWS;
        const float mean = ws[o] * invM;
        const float var  = ws[512 + o] * invM - mean * mean;
        const float sc   = gamma[o] * rsqrtf(var + 1e-5f);
        ws[1024 + o] = sc;
        ws[1536 + o] = beta[o] - mean * sc;
    }
}

// ---------------------------------------------------------------------------
// Kernel 4: in-place BN + multistep LIF (tau=2, v_th=1, hard reset)
// ---------------------------------------------------------------------------
__global__ __launch_bounds__(256) void bn_lif_kernel(
    float* __restrict__ Y, const float* __restrict__ ws)
{
    const int idx = blockIdx.x * 256 + threadIdx.x;   // < PER_T
    const int o   = idx & (NDIM - 1);
    const float scale = ws[1024 + o];
    const float shift = ws[1536 + o];
    float v = 0.f;
    #pragma unroll
    for (int t = 0; t < TSTEPS; ++t) {
        const long g  = (long)t * PER_T + idx;
        const float y = Y[g] * scale + shift;
        v += 0.5f * (y - v);
        const float s = (v >= 1.0f) ? 1.0f : 0.0f;
        Y[g] = s;
        v *= (1.0f - s);
    }
}

// ---------------------------------------------------------------------------
extern "C" void kernel_launch(void* const* d_in, const int* in_sizes, int n_in,
                              void* d_out, int out_size, void* d_ws, size_t ws_size,
                              hipStream_t stream)
{
    const float* x     = (const float*)d_in[0];
    const float* W     = (const float*)d_in[1];
    const float* gamma = (const float*)d_in[2];
    const float* beta  = (const float*)d_in[3];
    float* out = (float*)d_out;      // staged as y, finished as spikes
    float* ws  = (float*)d_ws;       // stats (8KB) + Wh/Wl planes (1MB)

    unsigned short* Wh = (unsigned short*)((char*)d_ws + WS_WH_OFF);
    unsigned short* Wl = Wh + (size_t)KDIM * NDIM;

    (void)hipMemsetAsync(ws, 0, 1024 * sizeof(float), stream);
    split_w_kernel<<<(KDIM * NDIM) / 256, 256, 0, stream>>>(W, Wh, Wl);

    dim3 gemmGrid(NDIM / BN, MROWS / BM);   // (4, 1024)
    gemm_bf16x3_wmma<<<gemmGrid, 256, 0, stream>>>(x, Wh, Wl, out);

    stats_kernel<<<MROWS / 128, 256, 0, stream>>>(out, ws);
    finalize_kernel<<<2, 256, 0, stream>>>(gamma, beta, ws);
    bn_lif_kernel<<<PER_T / 256, 256, 0, stream>>>(out, ws);
}
